// SecureAttention_45397804319070
// MI455X (gfx1250) — compile-verified, matrix operand order checked by reading, not run on hardware
//
#include <hip/hip_runtime.h>
#include <hip/hip_bf16.h>
#include <math.h>

typedef __attribute__((ext_vector_type(16))) _Float16 v16h;
typedef __attribute__((ext_vector_type(8)))  _Float16 v8h;
typedef __attribute__((ext_vector_type(8)))  float    v8f;
typedef int v4i_ __attribute__((vector_size(16)));   // matches builtin param

constexpr int B_  = 2;
constexpr int S_  = 1024;
constexpr int D_  = 1024;
constexpr int H_  = 16;
constexpr int HD_ = 64;
constexpr int BSD = B_ * S_ * D_;     // 2097152
constexpr int D2  = D_ * D_;          // 1048576
constexpr float MASKV = -10000.0f;
constexpr float SCALE = 0.125f;       // 1/sqrt(64)

#define WMMA_F16(a, b, c) \
  __builtin_amdgcn_wmma_f32_16x16x32_f16(false, (a), false, (b), (short)0, (c), false, false)

// ---------------- fragment load (works for global or LDS pointers) ----------
__device__ __forceinline__ v16h load_frag16(const _Float16* __restrict__ p) {
  v8h lo = *(const v8h*)(p);
  v8h hi = *(const v8h*)(p + 16);
  v16h r;
#pragma unroll
  for (int i = 0; i < 8; ++i) { r[i] = lo[i]; r[i + 8] = hi[i]; }
  return r;
}

// ---------------- async global -> LDS copy (16 bytes), guarded --------------
__device__ __forceinline__ void async_b128(const _Float16* g, _Float16* l) {
#if __has_builtin(__builtin_amdgcn_global_load_async_to_lds_b128)
  __builtin_amdgcn_global_load_async_to_lds_b128(
      (__attribute__((address_space(1))) v4i_*)(g),
      (__attribute__((address_space(3))) v4i_*)(l), 0, 0);
#else
  *(v8h*)l = *(const v8h*)g;              // synchronous fallback
#endif
}

__device__ __forceinline__ void wait_async_le(int pending) {
#if __has_builtin(__builtin_amdgcn_s_wait_asynccnt)
  if (pending) __builtin_amdgcn_s_wait_asynccnt(4);
  else         __builtin_amdgcn_s_wait_asynccnt(0);
#else
  if (pending) asm volatile("s_wait_asynccnt 0x4" ::: "memory");
  else         asm volatile("s_wait_asynccnt 0x0" ::: "memory");
#endif
}

// ---------------- direct-global WMMA core, register double-buffered ---------
// One wave computes a 16x64 strip of C. B columns are rows of a "rows-are-N"
// matrix (torch Linear weight W[n][k], K rows, or V^T rows).
__device__ __forceinline__ void zero4(v8f acc[4]) {
#pragma unroll
  for (int j = 0; j < 4; ++j)
#pragma unroll
    for (int e = 0; e < 8; ++e) acc[j][e] = 0.0f;
}

__device__ __forceinline__ void wmma_16x64(const _Float16* __restrict__ A, int lda,
                                           const _Float16* __restrict__ Bn, int ldb,
                                           int m0, int n0, int K, v8f acc[4]) {
  const int lane = threadIdx.x & 31;
  const int lr   = lane & 15;
  const int kb   = (lane < 16) ? 0 : 8;
  const _Float16* ap  = A  + (size_t)(m0 + lr) * lda + kb;
  const _Float16* bp0 = Bn + (size_t)(n0 + lr) * ldb + kb;
  const _Float16* bp1 = bp0 + (size_t)16 * ldb;
  const _Float16* bp2 = bp0 + (size_t)32 * ldb;
  const _Float16* bp3 = bp0 + (size_t)48 * ldb;
  v16h a  = load_frag16(ap);
  v16h b0 = load_frag16(bp0);
  v16h b1 = load_frag16(bp1);
  v16h b2 = load_frag16(bp2);
  v16h b3 = load_frag16(bp3);
  for (int k0 = 0; k0 < K; k0 += 32) {
    v16h an = a, c0 = b0, c1 = b1, c2 = b2, c3 = b3;
    if (k0 + 32 < K) {               // prefetch next fragments into fresh regs
      ap += 32; bp0 += 32; bp1 += 32; bp2 += 32; bp3 += 32;
      __builtin_prefetch(ap + 32, 0, 1);
      __builtin_prefetch(bp0 + 32, 0, 1);
      an = load_frag16(ap);
      c0 = load_frag16(bp0); c1 = load_frag16(bp1);
      c2 = load_frag16(bp2); c3 = load_frag16(bp3);
    }
    acc[0] = WMMA_F16(a, b0, acc[0]);
    acc[1] = WMMA_F16(a, b1, acc[1]);
    acc[2] = WMMA_F16(a, b2, acc[2]);
    acc[3] = WMMA_F16(a, b3, acc[3]);
    a = an; b0 = c0; b1 = c1; b2 = c2; b3 = c3;
  }
}

// ---------------- elementwise prep ------------------------------------------
__global__ void k_sum_convert(const float* __restrict__ h, _Float16* __restrict__ x16, int n) {
  int i = blockIdx.x * blockDim.x + threadIdx.x;
  if (i < n) x16[i] = (_Float16)(h[i] + h[i + n]);   // recon of the 2 shares
}

__global__ void k_convert(const float* __restrict__ w, _Float16* __restrict__ w16, int n) {
  int i = blockIdx.x * blockDim.x + threadIdx.x;
  if (i < n) w16[i] = (_Float16)w[i];
}

// ---------------- LDS-tiled GEMM: C[2048,1024] = X @ W.T (+bias) ------------
// Workgroup = 8 waves computes a 128x128 tile; K-step 32 halves, double
// buffered in LDS, filled with async global->LDS DMA (ASYNCcnt).
// mode 0: f16 out[b][h][s][d]   (Q, K projections)
// mode 1: f16 out[b][h][d][s]   (V^T)
// mode 3: f32 flat [row][col]   (output projection -> d_out)
constexpr int TS = 40;          // LDS row stride in halves (80B: conflict-free)
__global__ void k_gemm_lds(const _Float16* __restrict__ X, const _Float16* __restrict__ W,
                           const float* __restrict__ bias, _Float16* __restrict__ out16,
                           float* __restrict__ out32, int mode) {
  __shared__ _Float16 sA[2][128 * TS];
  __shared__ _Float16 sB[2][128 * TS];
  const int t = threadIdx.x;
  const int m_base = (blockIdx.x & 15) << 7;      // 16 m-blocks of 128
  const int n_base = (blockIdx.x >> 4) << 7;      // 8 n-blocks of 128
  // copy role: thread t moves one 16-half chunk of A and of B per stage
  const int cr = t >> 1;            // row 0..127
  const int cc = (t & 1) << 4;      // half-offset 0 or 16
  const _Float16* gA = X + (size_t)(m_base + cr) * D_ + cc;
  const _Float16* gB = W + (size_t)(n_base + cr) * D_ + cc;
  _Float16* lA = &sA[0][cr * TS + cc];
  _Float16* lB = &sB[0][cr * TS + cc];
  const int lbuf = 128 * TS;

  auto stage = [&](int buf, int k0) {
    async_b128(gA + k0,     lA + buf * lbuf);
    async_b128(gA + k0 + 8, lA + buf * lbuf + 8);
    async_b128(gB + k0,     lB + buf * lbuf);
    async_b128(gB + k0 + 8, lB + buf * lbuf + 8);
  };

  const int lane = t & 31, lr = lane & 15;
  const int kb = (lane < 16) ? 0 : 8;
  const int w16r = (t >> 5) << 4;           // wave's 16-row band
  v8f acc[8];
#pragma unroll
  for (int j = 0; j < 8; ++j)
#pragma unroll
    for (int e = 0; e < 8; ++e) acc[j][e] = 0.0f;

  constexpr int NK = D_ / 32;               // 32 K-steps
  stage(0, 0);
  for (int kt = 0; kt < NK; ++kt) {
    const int buf = kt & 1;
    const int more = (kt + 1 < NK);
    if (more) stage(buf ^ 1, (kt + 1) * 32);
    wait_async_le(more);                    // own copies for `buf` are done
    __syncthreads();                        // everyone's copies are done
    const _Float16* sa = &sA[buf][(w16r + lr) * TS + kb];
    v16h a = load_frag16(sa);
#pragma unroll
    for (int j = 0; j < 8; ++j) {
      v16h b = load_frag16(&sB[buf][(j * 16 + lr) * TS + kb]);
      acc[j] = WMMA_F16(a, b, acc[j]);
    }
    __syncthreads();                        // reads done before buf is refilled
  }

  const int rhi = (lane >= 16) ? 8 : 0;
#pragma unroll
  for (int j = 0; j < 8; ++j) {
    const int col = n_base + j * 16 + lr;
    const int h = col >> 6, d = col & 63;
    const float bb = bias[col];
#pragma unroll
    for (int r = 0; r < 8; ++r) {
      const int row = m_base + w16r + r + rhi;
      const int b = row >> 10, s = row & 1023;
      const float v = acc[j][r] + bb;
      if (mode == 3)      out32[((size_t)row << 10) + col] = v;
      else if (mode == 1) out16[((size_t)((b * H_ + h) * HD_ + d) << 10) + s] = (_Float16)v;
      else                out16[(((size_t)((b * H_ + h) << 10) + s) << 6) + d] = (_Float16)v;
    }
  }
}

// ---------------- scores = Q K^T * scale + additive mask --------------------
__global__ void k_scores(const _Float16* __restrict__ Qh, const _Float16* __restrict__ Kh,
                         const float* __restrict__ mask, _Float16* __restrict__ probs) {
  const int wid = (blockIdx.x * blockDim.x + threadIdx.x) >> 5;
  const int bh  = wid >> 10;
  const int rem = wid & 1023;
  const int m0 = (rem & 63) << 4;
  const int n0 = (rem >> 6) << 6;
  const _Float16* A  = Qh + ((size_t)bh << 16);   // bh * S * HD
  const _Float16* Bn = Kh + ((size_t)bh << 16);
  v8f acc[4]; zero4(acc);
  wmma_16x64(A, HD_, Bn, HD_, m0, n0, HD_, acc);
  const int lane = threadIdx.x & 31;
  const int lr = lane & 15, rhi = (lane >= 16) ? 8 : 0;
  const int b = bh >> 4;
  _Float16* prow = probs + ((size_t)bh << 20);    // bh * S * S
#pragma unroll
  for (int j = 0; j < 4; ++j) {
    const int col = n0 + j * 16 + lr;
    const float add = (1.0f - mask[(b << 10) + col]) * MASKV;
#pragma unroll
    for (int r = 0; r < 8; ++r) {
      const int row = m0 + r + rhi;
      prow[((size_t)row << 10) + col] = (_Float16)(acc[j][r] * SCALE + add);
    }
  }
}

// ---------------- row softmax over 1024 entries, one wave per row -----------
__global__ void k_softmax(_Float16* __restrict__ probs) {
  const int row  = (blockIdx.x * blockDim.x + threadIdx.x) >> 5;
  const int lane = threadIdx.x & 31;
  _Float16* p = probs + ((size_t)row << 10);
  float v[32];
  float mx = -3.4e38f;
#pragma unroll
  for (int i = 0; i < 32; ++i) { v[i] = (float)p[lane + (i << 5)]; mx = fmaxf(mx, v[i]); }
#pragma unroll
  for (int o = 16; o > 0; o >>= 1) mx = fmaxf(mx, __shfl_xor(mx, o, 32));
  float sum = 0.0f;
#pragma unroll
  for (int i = 0; i < 32; ++i) { v[i] = __expf(v[i] - mx); sum += v[i]; }
#pragma unroll
  for (int o = 16; o > 0; o >>= 1) sum += __shfl_xor(sum, o, 32);
  const float inv = 1.0f / sum;
#pragma unroll
  for (int i = 0; i < 32; ++i) p[lane + (i << 5)] = (_Float16)(v[i] * inv);
}

// ---------------- ctx = probs @ V  (B columns = rows of V^T) ----------------
__global__ void k_ctx(const _Float16* __restrict__ probs, const _Float16* __restrict__ Vt,
                      _Float16* __restrict__ ctx16) {
  const int wid = (blockIdx.x * blockDim.x + threadIdx.x) >> 5;
  const int bh = wid >> 6;
  const int m0 = (wid & 63) << 4;
  const _Float16* A  = probs + ((size_t)bh << 20);
  const _Float16* Bn = Vt    + ((size_t)bh << 16);
  v8f acc[4]; zero4(acc);
  wmma_16x64(A, S_, Bn, S_, m0, 0, S_, acc);
  const int lane = threadIdx.x & 31;
  const int lr = lane & 15, rhi = (lane >= 16) ? 8 : 0;
  const int b = bh >> 4, h = bh & 15;
#pragma unroll
  for (int j = 0; j < 4; ++j) {
    const int d = j * 16 + lr;
#pragma unroll
    for (int r = 0; r < 8; ++r) {
      const int s = m0 + r + rhi;
      ctx16[(((size_t)((b << 10) + s)) << 10) + (h << 6) + d] = (_Float16)acc[j][r];
    }
  }
}

// ---------------- launcher ---------------------------------------------------
extern "C" void kernel_launch(void* const* d_in, const int* in_sizes, int n_in,
                              void* d_out, int out_size, void* d_ws, size_t ws_size,
                              hipStream_t stream) {
  (void)in_sizes; (void)n_in; (void)out_size; (void)ws_size;
  const float* hid  = (const float*)d_in[0];
  const float* mask = (const float*)d_in[1];
  const float* wq   = (const float*)d_in[2];
  const float* bq   = (const float*)d_in[3];
  const float* wk   = (const float*)d_in[4];
  const float* bk   = (const float*)d_in[5];
  const float* wv   = (const float*)d_in[6];
  const float* bv   = (const float*)d_in[7];
  const float* wo   = (const float*)d_in[8];
  const float* bo   = (const float*)d_in[9];
  float* out = (float*)d_out;

  char* ws = (char*)d_ws;
  const size_t MB = 1u << 20;
  _Float16* x16   = (_Float16*)(ws + 0 * MB);
  _Float16* qh    = (_Float16*)(ws + 4 * MB);
  _Float16* kh    = (_Float16*)(ws + 8 * MB);
  _Float16* vt    = (_Float16*)(ws + 12 * MB);
  _Float16* ctx   = (_Float16*)(ws + 16 * MB);
  _Float16* w16q  = (_Float16*)(ws + 20 * MB);
  _Float16* w16k  = (_Float16*)(ws + 22 * MB);
  _Float16* w16v  = (_Float16*)(ws + 24 * MB);
  _Float16* w16o  = (_Float16*)(ws + 26 * MB);
  _Float16* probs = (_Float16*)(ws + 28 * MB);   // 64 MB

  k_sum_convert<<<(BSD + 255) / 256, 256, 0, stream>>>(hid, x16, BSD);
  k_convert<<<(D2 + 255) / 256, 256, 0, stream>>>(wq, w16q, D2);
  k_convert<<<(D2 + 255) / 256, 256, 0, stream>>>(wk, w16k, D2);
  k_convert<<<(D2 + 255) / 256, 256, 0, stream>>>(wv, w16v, D2);
  k_convert<<<(D2 + 255) / 256, 256, 0, stream>>>(wo, w16o, D2);

  // Q, K in [b][h][s][d]; V written transposed [b][h][d][s]
  k_gemm_lds<<<128, 256, 0, stream>>>(x16, w16q, bq, qh, nullptr, 0);
  k_gemm_lds<<<128, 256, 0, stream>>>(x16, w16k, bk, kh, nullptr, 0);
  k_gemm_lds<<<128, 256, 0, stream>>>(x16, w16v, bv, vt, nullptr, 1);

  k_scores <<<4096, 256, 0, stream>>>(qh, kh, mask, probs);
  k_softmax<<<4096, 256, 0, stream>>>(probs);
  k_ctx    <<<256, 256, 0, stream>>>(probs, vt, ctx);

  k_gemm_lds<<<128, 256, 0, stream>>>(ctx, w16o, bo, nullptr, out, 3);
}
